// NodeLayer_47261820125412
// MI455X (gfx1250) — matrix operand order repaired and checked by reading, not compiled, not run on hardware
//
#include <hip/hip_runtime.h>

// ---------------- constants ----------------
#define Bsz     16384
#define Udim    128
#define Hdim    512
#define HS      520          // padded bf16 row stride for h buffer
#define DTv     0.1f
#define LNEPS   1e-3f
#define BDIM    256
#define ROWS_PB 32           // rows per block
#define NE      16           // fp32 state elements per thread (32*128/256)

typedef __attribute__((ext_vector_type(16))) __bf16 v16bf;
typedef __attribute__((ext_vector_type(8)))  float  v8f;

__device__ __forceinline__ __bf16 f2bf(float f) { return (__bf16)f; }   // native v_cvt, RNE
__device__ __forceinline__ float  bf2f(__bf16 b) { return (float)b; }

// Position (lane*16 + elem) of element (row m, k-local kl) inside a 16x32
// bf16 A-fragment tile, contiguous k order (lane half h holds k = 16h + e).
__device__ __forceinline__ int apos(int m, int kl) {
    int half = (kl >> 4) & 1;               // kl / 16
    int lane = m + (half << 4);
    int e    = kl & 15;
    return lane * 16 + e;
}

__global__ void zero_reg_kernel(float* p) { *p = 0.0f; }

__global__ void __launch_bounds__(BDIM, 1)
ode_mlp_kernel(const float* __restrict__ x0,
               const float* __restrict__ W1g, const float* __restrict__ b1g,
               const float* __restrict__ gmg, const float* __restrict__ btg,
               const float* __restrict__ W2g, const float* __restrict__ b2g,
               float* __restrict__ out, float* __restrict__ regout)
{
    // B-fragment weight caches (bf16), tile-major: each tile = 32 lanes * 16 elems
    __shared__ __align__(32) __bf16 s_w1[4 * 32 * 32 * 16];   // 128 KB  (kb,nt) K=128,N=512
    __shared__ __align__(32) __bf16 s_w2[16 * 8 * 32 * 16];   // 128 KB  (kb,nt) K=512,N=128
    __shared__ __align__(32) __bf16 s_af[16 * 32 * 16];       // 16 KB: A-frags (GEMM1 uses first 4 tiles)
    __shared__ __align__(32) __bf16 s_hb[16 * HS];            // h (bf16) 16 rows
    __shared__ float s_kb[16 * Udim];                         // k output 16 rows
    __shared__ float s_mu[16], s_rs[16];
    __shared__ float s_regsum;

    const int t    = threadIdx.x;
    const int lane = t & 31;
    const int wv   = t >> 5;

    // ---- load + swizzle weights into LDS as bf16 B-fragments ----
    for (int idx = t; idx < 4 * 32 * 32 * 16; idx += BDIM) {
        int e = idx & 15, ln = (idx >> 4) & 31, tile = idx >> 9; // tile = kb*32+nt
        int nt = tile & 31, kb = tile >> 5;
        int K = kb * 32 + ((ln >> 4) << 4) + e;      // B layout: lanes16-31 hold K+16
        int N = nt * 16 + (ln & 15);
        s_w1[idx] = f2bf(W1g[K * Hdim + N]);
    }
    for (int idx = t; idx < 16 * 8 * 32 * 16; idx += BDIM) {
        int e = idx & 15, ln = (idx >> 4) & 31, tile = idx >> 9; // tile = kb*8+nt
        int nt = tile & 7, kb = tile >> 3;
        int K = kb * 32 + ((ln >> 4) << 4) + e;
        int N = nt * 16 + (ln & 15);
        s_w2[idx] = f2bf(W2g[K * Udim + N]);
    }
    if (t == 0) s_regsum = 0.0f;
    __syncthreads();

    // ---- persistent per-thread state: element g = p*2048 + jj*256 + t ----
    const int base = blockIdx.x * (ROWS_PB * Udim);
    float x[NE];
#pragma unroll
    for (int i = 0; i < NE; ++i)
        x[i] = x0[base + (i >> 3) * 2048 + (i & 7) * 256 + t];

    // ---- f(xb + c*kp) -> kout (kout may alias kp: each phase reads its slice first) ----
    auto eval_f = [&](const float* xb, const float* kp, float c, float* kout, bool do_reg) {
#pragma unroll
        for (int p = 0; p < 2; ++p) {                 // two 16-row phases
            // scatter stage input into GEMM1 A-fragments (bf16), fused x + c*k
#pragma unroll
            for (int jj = 0; jj < 8; ++jj) {
                int g = p * 2048 + jj * 256 + t;
                int m = (g >> 7) & 15, col = g & 127;
                float v = xb[p * 8 + jj] + c * kp[p * 8 + jj];
                s_af[(col >> 5) * 512 + apos(m, col & 31)] = f2bf(v);
            }
            __syncthreads();
            // GEMM1: h = x*W1 (+b1), each wave: 4 N-tiles, K=128 -> 4 WMMAs each
#pragma unroll
            for (int q = 0; q < 4; ++q) {
                int nt = wv * 4 + q;
                v8f acc = {};
#pragma unroll
                for (int kb = 0; kb < 4; ++kb) {
                    v16bf a = *(const v16bf*)(s_af + kb * 512 + lane * 16);
                    v16bf b = *(const v16bf*)(s_w1 + (kb * 32 + nt) * 512 + lane * 16);
                    acc = __builtin_amdgcn_wmma_f32_16x16x32_bf16(
                              false, a, false, b, (short)0, acc, false, false);
                }
                int n = nt * 16 + (lane & 15);
                float bb = b1g[n];
#pragma unroll
                for (int r = 0; r < 8; ++r) {
                    int m = (lane < 16) ? r : (r + 8);
                    s_hb[m * HS + n] = f2bf(acc[r] + bb);
                }
            }
            __syncthreads();
            // LayerNorm stats: 16 threads per row, in-wave shuffle reduction
            {
                int row = t >> 4, sg = t & 15;
                float s1 = 0.f, s2 = 0.f;
                for (int col = sg * 32; col < sg * 32 + 32; ++col) {
                    float v = bf2f(s_hb[row * HS + col]);
                    s1 += v; s2 += v * v;
                }
#pragma unroll
                for (int m = 1; m < 16; m <<= 1) {
                    s1 += __shfl_xor(s1, m, 32);
                    s2 += __shfl_xor(s2, m, 32);
                }
                if (sg == 0) {
                    float mu  = s1 * (1.0f / Hdim);
                    float var = s2 * (1.0f / Hdim) - mu * mu;
                    s_mu[row] = mu;
                    s_rs[row] = rsqrtf(var + LNEPS);
                }
            }
            __syncthreads();
            // normalize + ReLU -> GEMM2 A-fragments (bf16)
            for (int idx = t; idx < 16 * Hdim; idx += BDIM) {
                int m = idx >> 9, col = idx & 511;
                float v = bf2f(s_hb[m * HS + col]);
                v = (v - s_mu[m]) * s_rs[m] * gmg[col] + btg[col];
                v = fmaxf(v, 0.0f);
                s_af[(col >> 5) * 512 + apos(m, col & 31)] = f2bf(v);
            }
            __syncthreads();
            // GEMM2: k = h*W2 (+b2), each wave: 1 N-tile, K=512 -> 16 WMMAs
            {
                v8f acc = {};
#pragma unroll
                for (int kb = 0; kb < 16; ++kb) {
                    v16bf a = *(const v16bf*)(s_af + kb * 512 + lane * 16);
                    v16bf b = *(const v16bf*)(s_w2 + (kb * 8 + wv) * 512 + lane * 16);
                    acc = __builtin_amdgcn_wmma_f32_16x16x32_bf16(
                              false, a, false, b, (short)0, acc, false, false);
                }
                int n = wv * 16 + (lane & 15);
                float bb = b2g[n];
#pragma unroll
                for (int r = 0; r < 8; ++r) {
                    int m = (lane < 16) ? r : (r + 8);
                    s_kb[m * Udim + n] = acc[r] + bb;
                }
            }
            __syncthreads();
            if (do_reg) {   // softmax-weighted |pvf| per row, in-wave shuffle reductions
                int row = t >> 4, sg = t & 15;
                float mx = 0.0f;
                for (int col = sg * 8; col < sg * 8 + 8; ++col)
                    mx = fmaxf(mx, fabsf(s_kb[row * Udim + col]));
#pragma unroll
                for (int m = 1; m < 16; m <<= 1)
                    mx = fmaxf(mx, __shfl_xor(mx, m, 32));
                float e1 = 0.f, e2 = 0.f;
                for (int col = sg * 8; col < sg * 8 + 8; ++col) {
                    float a = fabsf(s_kb[row * Udim + col]);
                    float w = __expf(a - mx);
                    e1 += w; e2 += a * w;
                }
#pragma unroll
                for (int m = 1; m < 16; m <<= 1) {
                    e1 += __shfl_xor(e1, m, 32);
                    e2 += __shfl_xor(e2, m, 32);
                }
                if (sg == 0) atomicAdd(&s_regsum, e2 / e1);
            }
            // gather k back to registers (no trailing barrier needed: next write to
            // s_kb / s_af is separated by multiple barriers in the next phase/eval)
#pragma unroll
            for (int jj = 0; jj < 8; ++jj) {
                int g = p * 2048 + jj * 256 + t;
                kout[p * 8 + jj] = s_kb[((g >> 7) & 15) * Udim + (g & 127)];
            }
        }
    };

    // ---- RK4: 10 steps + 5 segments x 2 steps; reg-eval after steps 12,14,16,18,20 ----
    float kk[NE], av[NE];
    for (int step = 0; step < 20; ++step) {
#pragma unroll
        for (int i = 0; i < NE; ++i) { kk[i] = 0.0f; av[i] = 0.0f; }
        for (int s = 0; s < 4; ++s) {
            float c = (s == 0) ? 0.0f : ((s == 3) ? DTv : 0.5f * DTv);
            float w = (s == 1 || s == 2) ? 2.0f : 1.0f;
            eval_f(x, kk, c, kk, false);
#pragma unroll
            for (int i = 0; i < NE; ++i) av[i] += w * kk[i];
        }
#pragma unroll
        for (int i = 0; i < NE; ++i) x[i] += (DTv / 6.0f) * av[i];
        if (step >= 11 && ((step - 11) & 1) == 0) {   // segment end
            eval_f(x, kk, 0.0f, kk, true);
        }
    }

    // ---- outputs ----
#pragma unroll
    for (int i = 0; i < NE; ++i)
        out[base + (i >> 3) * 2048 + (i & 7) * 256 + t] = x[i];
    __syncthreads();
    if (t == 0)
        atomicAdd(regout, s_regsum * (1.0f / (5.0f * (float)Bsz)));  // REG_FACTOR=1, mean over 5*B
}

extern "C" void kernel_launch(void* const* d_in, const int* in_sizes, int n_in,
                              void* d_out, int out_size, void* d_ws, size_t ws_size,
                              hipStream_t stream) {
    (void)in_sizes; (void)n_in; (void)d_ws; (void)ws_size; (void)out_size;
    const float* x0 = (const float*)d_in[0];
    const float* W1 = (const float*)d_in[1];
    const float* b1 = (const float*)d_in[2];
    const float* gm = (const float*)d_in[3];
    const float* bt = (const float*)d_in[4];
    const float* W2 = (const float*)d_in[5];
    const float* b2 = (const float*)d_in[6];
    float* out  = (float*)d_out;
    float* regp = out + (size_t)Bsz * Udim;

    zero_reg_kernel<<<1, 1, 0, stream>>>(regp);
    ode_mlp_kernel<<<Bsz / ROWS_PB, BDIM, 0, stream>>>(x0, W1, b1, gm, bt, W2, b2, out, regp);
}
